// Block_7559142441551
// MI455X (gfx1250) — matrix operand order repaired and checked by reading, not compiled, not run on hardware
//
#include <hip/hip_runtime.h>
#include <hip/hip_bf16.h>
#include <cstdint>

// ---------------------------------------------------------------------------
// GPT-2 block on gfx1250 (MI455X). All matmuls via v_wmma_f32_16x16x32_f16.
// Double-buffered async global->LDS staging (ASYNCcnt) to overlap HBM/L2
// latency with the WMMA pipeline.
// B=4, T=2048, C=1024, H=16, hd=64. M = B*T = 8192 tokens.
// ---------------------------------------------------------------------------

typedef __attribute__((ext_vector_type(16))) _Float16 v16h;
typedef __attribute__((ext_vector_type(8)))  float    v8f;

#define BDIM 256

__device__ __forceinline__ v8f wmma32(v16h a, v16h b, v8f c) {
  return __builtin_amdgcn_wmma_f32_16x16x32_f16(
      /*neg_a=*/false, a, /*neg_b=*/false, b,
      /*c_mod=*/(short)0, c, /*reuse_a=*/false, /*reuse_b=*/false);
}

// Async 16B copy global -> LDS (per-lane addresses), tracked by ASYNCcnt.
__device__ __forceinline__ void async_copy_b128(unsigned lds_addr, const void* gptr) {
  asm volatile("global_load_async_to_lds_b128 %0, %1, off"
               :: "v"(lds_addr), "v"(gptr) : "memory");
}
__device__ __forceinline__ void wait_async0() {
  asm volatile("s_wait_asynccnt 0" ::: "memory");
}
__device__ __forceinline__ unsigned lds_off(const void* p) {
  return (unsigned)(uintptr_t)p;
}

// A-matrix fragment, 16x32 f16, row-major source with leading dim lda.
// Layout: lane m=lane&15; half=lane>>4; element s: K=s+16h (matches load_b_fragT).
// Contiguous runs of 8 halves -> compiler merges into b128 loads.
__device__ __forceinline__ v16h load_a_frag(const _Float16* A, int lda) {
  int lane = threadIdx.x & 31;
  int m  = lane & 15;
  int h8 = (lane >> 4) << 4;           // 0 or 16
  const _Float16* row = A + m * lda;
  v16h a;
#pragma unroll
  for (int r = 0; r < 8; ++r) {
    int kb = 2 * r + h8;
    a[2 * r]     = row[kb];
    a[2 * r + 1] = row[kb + 1];
  }
  return a;
}

// B-matrix fragment 32x16, source stored TRANSPOSED as Bp[n][k] (leading dim ldb).
// b[2r] = B[k=2r+16h][n] = Bp[n*ldb + 2r + 16h] -> 16 contiguous halves.
__device__ __forceinline__ v16h load_b_fragT(const _Float16* Bp, int ldb) {
  int lane = threadIdx.x & 31;
  int n   = lane & 15;
  int k16 = (lane >> 4) << 4;          // 0 or 16
  const _Float16* col = Bp + n * ldb + k16;
  v16h b;
#pragma unroll
  for (int r = 0; r < 8; ++r) {
    b[2 * r]     = col[2 * r];
    b[2 * r + 1] = col[2 * r + 1];
  }
  return b;
}

// ---------------------------------------------------------------------------
// Weight fp32 [K][N] -> f16 transposed Wt[N][K]
// ---------------------------------------------------------------------------
__global__ void convert_wt_kernel(const float* __restrict__ w,
                                  _Float16* __restrict__ wt, int K, int N) {
  int idx = blockIdx.x * BDIM + threadIdx.x;
  if (idx >= K * N) return;
  int n = idx / K;
  int k = idx - n * K;
  wt[idx] = (_Float16)w[(size_t)k * N + n];
}

// ---------------------------------------------------------------------------
// LayerNorm: one block per token row of 1024, output f16
// ---------------------------------------------------------------------------
__global__ __launch_bounds__(BDIM) void ln_kernel(const float* __restrict__ x,
                                                  const float* __restrict__ w,
                                                  const float* __restrict__ bia,
                                                  _Float16* __restrict__ out) {
  int row = blockIdx.x;
  int tid = threadIdx.x;
  const float4 v  = ((const float4*)(x + (size_t)row * 1024))[tid];
  float s  = v.x + v.y + v.z + v.w;
  float ss = v.x * v.x + v.y * v.y + v.z * v.z + v.w * v.w;
#pragma unroll
  for (int off = 1; off < 32; off <<= 1) {
    s  += __shfl_xor(s, off, 32);
    ss += __shfl_xor(ss, off, 32);
  }
  __shared__ float red[2][8];
  if ((tid & 31) == 0) { red[0][tid >> 5] = s; red[1][tid >> 5] = ss; }
  __syncthreads();
  float sum = 0.f, sumsq = 0.f;
#pragma unroll
  for (int i = 0; i < 8; ++i) { sum += red[0][i]; sumsq += red[1][i]; }
  const float mu   = sum * (1.0f / 1024.0f);
  const float var  = sumsq * (1.0f / 1024.0f) - mu * mu;
  const float rstd = rsqrtf(var + 1e-5f);
  const float4 wv = ((const float4*)w)[tid];
  const float4 bv = ((const float4*)bia)[tid];
  _Float16* o = out + (size_t)row * 1024 + tid * 4;
  o[0] = (_Float16)((v.x - mu) * rstd * wv.x + bv.x);
  o[1] = (_Float16)((v.y - mu) * rstd * wv.y + bv.y);
  o[2] = (_Float16)((v.z - mu) * rstd * wv.z + bv.z);
  o[3] = (_Float16)((v.w - mu) * rstd * wv.w + bv.w);
}

// ---------------------------------------------------------------------------
// Tiled GEMM: out[M,N] = A[M,K](f16) @ Wt[N,K](f16)^T + bias, epilogue by MODE.
// MODE 0: f16 out; MODE 1: f16 out + tanh-GELU; MODE 2: f32 out + residual.
// Block tile 128x128, 8 waves (2Mx4N), wave tile 64x32 (4x2 WMMA), K-step 32.
// Double-buffered async staging: next K-slice copies in flight during WMMAs.
// ---------------------------------------------------------------------------
#define ASTRIDE 40   // 32 + 8 pad halves: 20-dword row stride -> conflict-free
#define BSTRIDE 40

template <int MODE>
__global__ __launch_bounds__(BDIM) void gemm_kernel(
    const _Float16* __restrict__ A, const _Float16* __restrict__ Wt,
    const float* __restrict__ bias, const float* __restrict__ residual,
    void* __restrict__ outp, int M, int N, int K) {
  __shared__ _Float16 As[2][128 * ASTRIDE];
  __shared__ _Float16 Bs[2][128 * BSTRIDE];
  const int tid  = threadIdx.x;
  const int m0   = blockIdx.y * 128;
  const int n0   = blockIdx.x * 128;
  const int wave = tid >> 5;
  const int wm   = wave >> 2;            // 0..1 -> M offset wm*64
  const int wn   = wave & 3;             // 0..3 -> N offset wn*32
  const int lane = tid & 31;
  const int ln   = lane & 15;
  const int lh   = lane >> 4;

  // Stage one 128x32 K-slice of A and B into buffer `buf` (async, no wait).
  auto stage = [&](int buf, int k0) {
    const unsigned asb = lds_off(&As[buf][0]);
    const unsigned bsb = lds_off(&Bs[buf][0]);
#pragma unroll
    for (int i = 0; i < 2; ++i) {
      int t   = tid + i * BDIM;            // 0..511
      int row = t >> 2;                    // 0..127
      int q   = t & 3;                     // 16B quarter of the 64B row
      async_copy_b128(asb + row * (ASTRIDE * 2) + q * 16,
                      &A[(size_t)(m0 + row) * K + k0 + q * 8]);
      async_copy_b128(bsb + row * (BSTRIDE * 2) + q * 16,
                      &Wt[(size_t)(n0 + row) * K + k0 + q * 8]);
    }
  };

  v8f acc[4][2];
#pragma unroll
  for (int i = 0; i < 4; ++i)
#pragma unroll
    for (int j = 0; j < 2; ++j) acc[i][j] = (v8f){0, 0, 0, 0, 0, 0, 0, 0};

  stage(0, 0);
  wait_async0();
  __syncthreads();

  int cur = 0;
  for (int k0 = 0; k0 < K; k0 += 32) {
    const int nxt = cur ^ 1;
    if (k0 + 32 < K) stage(nxt, k0 + 32);  // prefetch next slice (in flight)

    v16h af[4], bf[2];
#pragma unroll
    for (int mt = 0; mt < 4; ++mt)
      af[mt] = load_a_frag(&As[cur][(wm * 64 + mt * 16) * ASTRIDE], ASTRIDE);
#pragma unroll
    for (int nt = 0; nt < 2; ++nt)
      bf[nt] = load_b_fragT(&Bs[cur][(wn * 32 + nt * 16) * BSTRIDE], BSTRIDE);
#pragma unroll
    for (int mt = 0; mt < 4; ++mt)
#pragma unroll
      for (int nt = 0; nt < 2; ++nt)
        acc[mt][nt] = wmma32(af[mt], bf[nt], acc[mt][nt]);

    wait_async0();       // next-slice copies complete (this wave's)
    __syncthreads();     // + all waves done reading `cur`
    cur = nxt;
  }

#pragma unroll
  for (int nt = 0; nt < 2; ++nt) {
    const int col = n0 + wn * 32 + nt * 16 + ln;
    const float bv = bias[col];
#pragma unroll
    for (int mt = 0; mt < 4; ++mt) {
#pragma unroll
      for (int r = 0; r < 8; ++r) {
        const int row = m0 + wm * 64 + mt * 16 + r + 8 * lh;
        float v = acc[mt][nt][r] + bv;
        const size_t idx = (size_t)row * N + col;
        if (MODE == 1) {
          const float c0 = 0.7978845608028654f;  // sqrt(2/pi)
          float u = c0 * (v + 0.044715f * v * v * v);
          v = 0.5f * v * (1.0f + tanhf(u));
        }
        if (MODE == 2)
          ((float*)outp)[idx] = residual[idx] + v;
        else
          ((_Float16*)outp)[idx] = (_Float16)v;
      }
    }
  }
}

// ---------------------------------------------------------------------------
// Flash attention. qkv f16 [8192][3072] (q|k|v each 1024 = 16 heads * 64).
// Grid: (T/128, H, B). Block: 8 waves; each wave owns 16 query rows.
// Double-buffered chunks: K staged async; V global loads issued pre-compute
// into registers, transposed into Vt[feat][key] LDS post-compute.
// ---------------------------------------------------------------------------
#define KVSTRIDE 72   // K chunk: 64 + 8 pad halves (144B rows, 16B aligned)
#define VTSTRIDE 36   // Vt: 32 keys + 4 pad halves, conflict-free frag reads
#define PSTRIDE  40

__global__ __launch_bounds__(BDIM) void attn_kernel(
    const _Float16* __restrict__ qkv, _Float16* __restrict__ y) {
  __shared__ _Float16 Ks[2][32 * KVSTRIDE];
  __shared__ _Float16 Vt[2][64 * VTSTRIDE];
  __shared__ _Float16 Ps[8][16 * PSTRIDE];

  const int tid  = threadIdx.x;
  const int wave = tid >> 5;
  const int lane = tid & 31;
  const int ln   = lane & 15;
  const int lh   = lane >> 4;
  const int b    = blockIdx.z;
  const int hh   = blockIdx.y;
  const int q0   = blockIdx.x * 128 + wave * 16;
  const size_t tok0 = (size_t)b * 2048;

  // K chunk: 32 keys x 128B rows = 256 x 16B async copies (1 per thread).
  auto stage_k = [&](int buf, int j0) {
    const unsigned ksb = lds_off(&Ks[buf][0]);
    int row = tid >> 3;                  // 0..31
    int q   = tid & 7;                   // 16B piece of the 128B row
    async_copy_b128(ksb + row * (KVSTRIDE * 2) + q * 16,
                    &qkv[(tok0 + j0 + row) * 3072 + 1024 + hh * 64 + q * 8]);
  };
  // V chunk: 2x2 micro-transpose; load phase (global->regs) ...
  auto load_v = [&](int j0, uint32_t* vr) {
#pragma unroll
    for (int i = 0; i < 2; ++i) {
      int t  = tid + i * BDIM;             // 0..511
      int ff = (t & 31) << 1;              // feat pair 0..62
      int kk = (t >> 5) << 1;              // key pair 0..30
      const _Float16* vg = qkv + (tok0 + j0 + kk) * 3072 + 2048 + hh * 64 + ff;
      vr[2 * i]     = *(const uint32_t*)vg;          // V[kk][ff..ff+1]
      vr[2 * i + 1] = *(const uint32_t*)(vg + 3072); // V[kk+1][ff..ff+1]
    }
  };
  // ... and store phase (regs->LDS, transposed).
  auto store_v = [&](int buf, const uint32_t* vr) {
#pragma unroll
    for (int i = 0; i < 2; ++i) {
      int t  = tid + i * BDIM;
      int ff = (t & 31) << 1;
      int kk = (t >> 5) << 1;
      union { uint32_t u; _Float16 h[2]; } r0, r1, c0, c1;
      r0.u = vr[2 * i];
      r1.u = vr[2 * i + 1];
      c0.h[0] = r0.h[0]; c0.h[1] = r1.h[0];
      c1.h[0] = r0.h[1]; c1.h[1] = r1.h[1];
      *(uint32_t*)&Vt[buf][ff * VTSTRIDE + kk]       = c0.u;
      *(uint32_t*)&Vt[buf][(ff + 1) * VTSTRIDE + kk] = c1.u;
    }
  };

  // Q fragments (feats 0..31 and 32..63), loaded once from global.
  const _Float16* qbase = qkv + (tok0 + q0) * 3072 + hh * 64;
  const v16h qa0 = load_a_frag(qbase, 3072);
  const v16h qa1 = load_a_frag(qbase + 32, 3072);

  float mst[8], lst[8];
#pragma unroll
  for (int r = 0; r < 8; ++r) { mst[r] = -1e30f; lst[r] = 0.0f; }
  v8f o[4];
#pragma unroll
  for (int nt = 0; nt < 4; ++nt) o[nt] = (v8f){0, 0, 0, 0, 0, 0, 0, 0};

  // Prologue: stage chunk 0.
  {
    uint32_t vr[4];
    stage_k(0, 0);
    load_v(0, vr);
    store_v(0, vr);
    wait_async0();
    __syncthreads();
  }

  int cur = 0;
  for (int j0 = 0; j0 < 2048; j0 += 32) {
    const int nxt = cur ^ 1;
    const bool pf = (j0 + 32) < 2048;
    uint32_t vr[4];
    if (pf) {
      stage_k(nxt, j0 + 32);     // async, in flight during compute
      load_v(j0 + 32, vr);       // global loads in flight during compute
    }

    // S tiles: keys j0..j0+15 (s0) and j0+16..j0+31 (s1), Kdim=64 (2 WMMA each).
    v8f s0 = (v8f){0, 0, 0, 0, 0, 0, 0, 0};
    v8f s1 = (v8f){0, 0, 0, 0, 0, 0, 0, 0};
    s0 = wmma32(qa0, load_b_fragT(&Ks[cur][0], KVSTRIDE), s0);
    s0 = wmma32(qa1, load_b_fragT(&Ks[cur][32], KVSTRIDE), s0);
    s1 = wmma32(qa0, load_b_fragT(&Ks[cur][16 * KVSTRIDE], KVSTRIDE), s1);
    s1 = wmma32(qa1, load_b_fragT(&Ks[cur][16 * KVSTRIDE + 32], KVSTRIDE), s1);

    float p0[8], p1[8], alpha[8];
#pragma unroll
    for (int r = 0; r < 8; ++r) {
      float a_ = s0[r] * 0.125f;                 // 1/sqrt(64)
      float b_ = s1[r] * 0.125f;
      float rm = fmaxf(a_, b_);
      rm = fmaxf(rm, __shfl_xor(rm, 1, 32));
      rm = fmaxf(rm, __shfl_xor(rm, 2, 32));
      rm = fmaxf(rm, __shfl_xor(rm, 4, 32));
      rm = fmaxf(rm, __shfl_xor(rm, 8, 32));
      float mn = fmaxf(mst[r], rm);
      alpha[r] = __expf(mst[r] - mn);
      float e0 = __expf(a_ - mn);
      float e1 = __expf(b_ - mn);
      float rs = e0 + e1;
      rs += __shfl_xor(rs, 1, 32);
      rs += __shfl_xor(rs, 2, 32);
      rs += __shfl_xor(rs, 4, 32);
      rs += __shfl_xor(rs, 8, 32);
      lst[r] = lst[r] * alpha[r] + rs;
      mst[r] = mn;
      p0[r] = e0;
      p1[r] = e1;
    }
#pragma unroll
    for (int nt = 0; nt < 4; ++nt)
#pragma unroll
      for (int r = 0; r < 8; ++r) o[nt][r] *= alpha[r];

    // Stage P (16x32, C-layout -> row-major LDS) and reload as A fragment.
    _Float16* pp = Ps[wave];
#pragma unroll
    for (int r = 0; r < 8; ++r) {
      pp[(r + 8 * lh) * PSTRIDE + ln]      = (_Float16)p0[r];
      pp[(r + 8 * lh) * PSTRIDE + ln + 16] = (_Float16)p1[r];
    }
    const v16h pa = load_a_frag(pp, PSTRIDE);   // same-wave LDS ops are in-order

    // O += P @ V : Vt gives contiguous B fragments per 16-feat tile.
#pragma unroll
    for (int nt = 0; nt < 4; ++nt)
      o[nt] = wmma32(pa, load_b_fragT(&Vt[cur][(nt * 16) * VTSTRIDE], VTSTRIDE), o[nt]);

    if (pf) store_v(nxt, vr);    // regs (already landed) -> LDS transpose
    wait_async0();               // K-chunk copies complete (this wave's)
    __syncthreads();             // + all waves done reading `cur`
    cur = nxt;
  }

#pragma unroll
  for (int r = 0; r < 8; ++r) lst[r] = 1.0f / lst[r];
#pragma unroll
  for (int nt = 0; nt < 4; ++nt) {
#pragma unroll
    for (int r = 0; r < 8; ++r) {
      const int row = q0 + r + 8 * lh;
      const int col = hh * 64 + nt * 16 + ln;
      y[(tok0 + row) * 1024 + col] = (_Float16)(o[nt][r] * lst[r]);
    }
  }
}

// ---------------------------------------------------------------------------
// Host-side orchestration
// ---------------------------------------------------------------------------
extern "C" void kernel_launch(void* const* d_in, const int* in_sizes, int n_in,
                              void* d_out, int out_size, void* d_ws, size_t ws_size,
                              hipStream_t stream) {
  (void)in_sizes; (void)n_in; (void)out_size; (void)ws_size;
  const float* x      = (const float*)d_in[0];
  // d_in[1] = t (unused)
  const float* ln1_w  = (const float*)d_in[2];
  const float* ln1_b  = (const float*)d_in[3];
  const float* attn_w = (const float*)d_in[4];
  const float* attn_b = (const float*)d_in[5];
  const float* proj_w = (const float*)d_in[6];
  const float* proj_b = (const float*)d_in[7];
  const float* ln2_w  = (const float*)d_in[8];
  const float* ln2_b  = (const float*)d_in[9];
  const float* fc_w   = (const float*)d_in[10];
  const float* fc_b   = (const float*)d_in[11];
  const float* fc2_w  = (const float*)d_in[12];
  const float* fc2_b  = (const float*)d_in[13];

  const int M = 8192;                     // B*T tokens
  char* ws = (char*)d_ws;
  // Workspace layout (~152 MB total); qkv and fc-activation share one region.
  _Float16* attn_wt = (_Float16*)(ws + 0);                 //  6 MB  [3072][1024]
  _Float16* proj_wt = (_Float16*)(ws + 6291456);           //  2 MB  [1024][1024]
  _Float16* fc_wt   = (_Float16*)(ws + 8388608);           //  8 MB  [4096][1024]
  _Float16* fc2_wt  = (_Float16*)(ws + 16777216);          //  8 MB  [1024][4096]
  _Float16* qkvbuf  = (_Float16*)(ws + 25165824);          // 48 MB  [8192][3072]
  _Float16* actbuf  = (_Float16*)(ws + 25165824);          // 64 MB  [8192][4096] (aliases qkv)
  _Float16* hbuf    = (_Float16*)(ws + 92274688);          // 16 MB  [8192][1024]
  _Float16* ybuf    = (_Float16*)(ws + 109051904);         // 16 MB  [8192][1024]
  float*    x1      = (float*)(ws + 125829120);            // 32 MB  [8192][1024]

  // Weights -> f16, transposed to [N][K].
  convert_wt_kernel<<<(1024 * 3072 + BDIM - 1) / BDIM, BDIM, 0, stream>>>(attn_w, attn_wt, 1024, 3072);
  convert_wt_kernel<<<(1024 * 1024 + BDIM - 1) / BDIM, BDIM, 0, stream>>>(proj_w, proj_wt, 1024, 1024);
  convert_wt_kernel<<<(1024 * 4096 + BDIM - 1) / BDIM, BDIM, 0, stream>>>(fc_w,   fc_wt,   1024, 4096);
  convert_wt_kernel<<<(4096 * 1024 + BDIM - 1) / BDIM, BDIM, 0, stream>>>(fc2_w,  fc2_wt,  4096, 1024);

  // LN1 -> h (f16)
  ln_kernel<<<M, BDIM, 0, stream>>>(x, ln1_w, ln1_b, hbuf);
  // qkv = h @ attn_w + attn_b  (f16 out)
  gemm_kernel<0><<<dim3(3072 / 128, M / 128), BDIM, 0, stream>>>(
      hbuf, attn_wt, attn_b, nullptr, qkvbuf, M, 3072, 1024);
  // attention -> y (f16)
  attn_kernel<<<dim3(16, 16, 4), BDIM, 0, stream>>>(qkvbuf, ybuf);
  // x1 = x + y @ proj_w + proj_b  (f32 out)
  gemm_kernel<2><<<dim3(1024 / 128, M / 128), BDIM, 0, stream>>>(
      ybuf, proj_wt, proj_b, x, x1, M, 1024, 1024);
  // LN2 -> h (f16, reuse)
  ln_kernel<<<M, BDIM, 0, stream>>>(x1, ln2_w, ln2_b, hbuf);
  // act = gelu(h @ fc_w + fc_b)  (f16 out)
  gemm_kernel<1><<<dim3(4096 / 128, M / 128), BDIM, 0, stream>>>(
      hbuf, fc_wt, fc_b, nullptr, actbuf, M, 4096, 1024);
  // out = x1 + act @ fc2_w + fc2_b  (f32 out)
  gemm_kernel<2><<<dim3(1024 / 128, M / 128), BDIM, 0, stream>>>(
      actbuf, fc2_wt, fc2_b, x1, d_out, M, 1024, 4096);
}
